// RegionResidualCalibration_7430293422638
// MI455X (gfx1250) — matrix operand order repaired and checked by reading, not compiled, not run on hardware
//
#include <hip/hip_runtime.h>

typedef __attribute__((ext_vector_type(16))) _Float16 v16h;
typedef __attribute__((ext_vector_type(8)))  _Float16 v8h;
typedef __attribute__((ext_vector_type(8)))  float    v8f;
typedef __attribute__((ext_vector_type(4)))  float    v4f;

// Problem sizes fixed by the reference.
constexpr int kB = 1024;      // users
constexpr int kS = 200;       // seq len
constexpr int kL = 50000;     // POIs
constexpr int kD = 128;       // embedding dim (= GEMM K)
constexpr int kRecent = 10;

constexpr int BT_STRIDE = 136; // halves per LDS row: 272B rows -> bank-conflict-free b128 reads

// ---------------------------------------------------------------------------
// Kernel 1: user_pref[b][d] = masked mean (f32) of region embeddings of last-10
// POIs; store rounded-to-f16 copy for the WMMA GEMM.
// ---------------------------------------------------------------------------
__global__ void __launch_bounds__(128) region_pref_kernel(
    const int* __restrict__ user_seq,        // [B,S]
    const int* __restrict__ user_seq_len,    // [B]
    const int* __restrict__ poi_region_id,   // [L]
    const float* __restrict__ region_emb,    // [R,D]
    _Float16* __restrict__ pref_h)           // [B,D] out (ws)
{
  const int b = blockIdx.x;
  const int d = threadIdx.x;                 // 0..127
  const int len = user_seq_len[b];

  float acc = 0.0f;
  int cnt = 0;
#pragma unroll
  for (int j = 0; j < kRecent; ++j) {
    int pos = len - kRecent + j;
    if (pos >= 0) {
      int p = pos > (kS - 1) ? (kS - 1) : pos;
      int poi = user_seq[b * kS + p];
      int rid = poi_region_id[poi];
      acc += region_emb[rid * kD + d];
      ++cnt;
    }
  }
  float denom = (cnt > 0) ? (float)cnt : 1.0f;
  pref_h[b * kD + d] = (_Float16)(acc / denom);
}

// ---------------------------------------------------------------------------
// Kernel 2: out[m][n] = pred[m][n] + alpha * dot(pref[m], region_emb[rid[n]])
// Tiled f16 WMMA GEMM: M=1024, N=50000, K=128.
// Block: 256 threads = 8 waves. Block tile: 128 rows (M) x 32 cols (N)
//   -> blocks cover full 128B cachelines of pred/out.
// Wave w: one 16-row M-tile, two 16-col N-tiles sharing the A fragment
//   -> 8 x v_wmma_f32_16x16x32_f16 per wave (K=128).
// ---------------------------------------------------------------------------
__global__ void __launch_bounds__(256) region_score_kernel(
    const float* __restrict__ pred,          // [B,L]
    const int* __restrict__ poi_region_id,   // [L]
    const float* __restrict__ region_emb,    // [R,D]
    const _Float16* __restrict__ pref_h,     // [B,D]
    const float* __restrict__ alpha_p,       // [1]
    float* __restrict__ out)                 // [B,L]
{
  __shared__ _Float16 Bt[32 * BT_STRIDE];    // r_emb^T tile: [n][k], padded rows

  const int n0 = blockIdx.x * 32;            // 1563 blocks; last covers 16 cols
  const int m0 = blockIdx.y * 128;           // 8 exact M blocks
  const int t  = threadIdx.x;

  // ---- Gather + convert 32 rows of region_emb into LDS (f32 -> f16) ----
  {
    const int n   = t >> 3;                  // 0..31
    const int seg = (t & 7) * 16;            // k chunk of 16
    if (n0 + n < kL) {
      const int rid = poi_region_id[n0 + n];
      const float* src = region_emb + rid * kD + seg;
      v4f f0 = *(const v4f*)(src);
      v4f f1 = *(const v4f*)(src + 4);
      v4f f2 = *(const v4f*)(src + 8);
      v4f f3 = *(const v4f*)(src + 12);
      v8h h0, h1;
      h0[0] = (_Float16)f0[0]; h0[1] = (_Float16)f0[1];
      h0[2] = (_Float16)f0[2]; h0[3] = (_Float16)f0[3];
      h0[4] = (_Float16)f1[0]; h0[5] = (_Float16)f1[1];
      h0[6] = (_Float16)f1[2]; h0[7] = (_Float16)f1[3];
      h1[0] = (_Float16)f2[0]; h1[1] = (_Float16)f2[1];
      h1[2] = (_Float16)f2[2]; h1[3] = (_Float16)f2[3];
      h1[4] = (_Float16)f3[0]; h1[5] = (_Float16)f3[1];
      h1[6] = (_Float16)f3[2]; h1[7] = (_Float16)f3[3];
      *(v8h*)(&Bt[n * BT_STRIDE + seg])     = h0;  // 16B aligned
      *(v8h*)(&Bt[n * BT_STRIDE + seg + 8]) = h1;
    }
  }
  __syncthreads();

  const int wave     = t >> 5;               // 0..7 -> M tile
  const int lane     = t & 31;
  const int half_sel = lane >> 4;            // 0: lanes 0-15, 1: lanes 16-31
  const int mn       = lane & 15;            // A row / B,C,D column within tile

  const _Float16* arow = pref_h + (m0 + wave * 16 + mn) * kD;
  const int koffA = half_sel * 16;           // A k order used here (contiguous): K = koffA..koffA+15
  const int koffB = half_sel * 16;           // B k order used here (contiguous): K = koffB..koffB+15

  const bool has2 = (n0 + 16) < kL;          // second N-tile in range? (uniform)

  // Prefetch this wave's pred rows (gfx1250 global_prefetch path).
  {
    int row = m0 + wave * 16 + half_sel * 8;
    __builtin_prefetch(&pred[row * kL + n0 + mn], 0, 0);
  }

  v8f acc0 = {};
  v8f acc1 = {};
#pragma unroll
  for (int kb = 0; kb < kD; kb += 32) {
    // A fragment: two contiguous 16B global loads per lane (layout-exact),
    // shared by both N-tiles.
    v8h alo = *(const v8h*)(arow + kb + koffA);
    v8h ahi = *(const v8h*)(arow + kb + 8 + koffA);
    v16h a = __builtin_shufflevector(alo, ahi,
        0, 1, 2, 3, 4, 5, 6, 7, 8, 9, 10, 11, 12, 13, 14, 15);

    // B fragments: 16 contiguous halves from LDS per tile (ds_load_b128 x2).
    const _Float16* bp0 = &Bt[mn * BT_STRIDE + kb + koffB];
    const _Float16* bp1 = &Bt[(mn + 16) * BT_STRIDE + kb + koffB];
    v8h b0lo = *(const v8h*)(bp0);
    v8h b0hi = *(const v8h*)(bp0 + 8);
    v16h bm0 = __builtin_shufflevector(b0lo, b0hi,
        0, 1, 2, 3, 4, 5, 6, 7, 8, 9, 10, 11, 12, 13, 14, 15);
    v8h b1lo = *(const v8h*)(bp1);
    v8h b1hi = *(const v8h*)(bp1 + 8);
    v16h bm1 = __builtin_shufflevector(b1lo, b1hi,
        0, 1, 2, 3, 4, 5, 6, 7, 8, 9, 10, 11, 12, 13, 14, 15);

    acc0 = __builtin_amdgcn_wmma_f32_16x16x32_f16(
        false, a, false, bm0, (short)0, acc0, false, false);
    acc1 = __builtin_amdgcn_wmma_f32_16x16x32_f16(
        false, a, false, bm1, (short)0, acc1, false, false);
  }

  const float alpha = *alpha_p;

  // C/D layout: VGPR r -> (lanes 0-15: M=r, lanes 16-31: M=8+r), N = lane&15.
#pragma unroll
  for (int r = 0; r < 8; ++r) {
    int row = m0 + wave * 16 + half_sel * 8 + r;
    int idx = row * kL + n0 + mn;            // max ~51.2M, fits int32
    out[idx] = pred[idx] + alpha * acc0[r];
  }
  if (has2) {
#pragma unroll
    for (int r = 0; r < 8; ++r) {
      int row = m0 + wave * 16 + half_sel * 8 + r;
      int idx = row * kL + n0 + 16 + mn;
      out[idx] = pred[idx] + alpha * acc1[r];
    }
  }
}

// ---------------------------------------------------------------------------
extern "C" void kernel_launch(void* const* d_in, const int* in_sizes, int n_in,
                              void* d_out, int out_size, void* d_ws, size_t ws_size,
                              hipStream_t stream) {
  const float* pred          = (const float*)d_in[0];
  const int*   user_seq      = (const int*)d_in[1];
  const int*   user_seq_len  = (const int*)d_in[2];
  const int*   poi_region_id = (const int*)d_in[3];
  const float* region_emb    = (const float*)d_in[4];
  const float* alpha         = (const float*)d_in[5];
  float*       out           = (float*)d_out;

  _Float16* pref_h = (_Float16*)d_ws;        // 1024*128*2 = 256 KB scratch

  region_pref_kernel<<<kB, kD, 0, stream>>>(
      user_seq, user_seq_len, poi_region_id, region_emb, pref_h);

  dim3 grid((kL + 31) / 32, kB / 128);       // (1563, 8)
  region_score_kernel<<<grid, 256, 0, stream>>>(
      pred, poi_region_id, region_emb, pref_h, alpha, out);
}
